// Mask2FormerAttention_48799418417544
// MI455X (gfx1250) — matrix operand order repaired and checked by reading, not compiled, not run on hardware
//
#include <hip/hip_runtime.h>

typedef __attribute__((ext_vector_type(16))) __bf16 v16bf;
typedef __attribute__((ext_vector_type(8)))  float  v8f;

union BF16x16 { v16bf v; uint4 q[2]; };
union BF16x8  { __bf16 h[8]; uint4 q; };

constexpr int D_DIM  = 1024;
constexpr int N_HEAD = 16;
constexpr int HD     = 64;
constexpr int T_LEN  = 1024;
constexpr int S_LEN  = 4096;
constexpr int BATCH  = 2;

constexpr int BM = 128, BN = 128, BK = 32;
constexpr int LDT = 40;   // LDS tile row stride (bf16 elems), 80B = 16B-aligned segments

__device__ __forceinline__ void cvt_store16(__bf16* dst, float4 f0, float4 f1,
                                            float4 f2, float4 f3) {
    BF16x8 h0, h1;
    h0.h[0]=(__bf16)f0.x; h0.h[1]=(__bf16)f0.y; h0.h[2]=(__bf16)f0.z; h0.h[3]=(__bf16)f0.w;
    h0.h[4]=(__bf16)f1.x; h0.h[5]=(__bf16)f1.y; h0.h[6]=(__bf16)f1.z; h0.h[7]=(__bf16)f1.w;
    h1.h[0]=(__bf16)f2.x; h1.h[1]=(__bf16)f2.y; h1.h[2]=(__bf16)f2.z; h1.h[3]=(__bf16)f2.w;
    h1.h[4]=(__bf16)f3.x; h1.h[5]=(__bf16)f3.y; h1.h[6]=(__bf16)f3.z; h1.h[7]=(__bf16)f3.w;
    *reinterpret_cast<uint4*>(dst)     = h0.q;
    *reinterpret_cast<uint4*>(dst + 8) = h1.q;
}

// C = (X [+ Xadd]) @ W^T, bias, scale.
// mode 0/1: out bf16 [(b*H+h), seqlen, HD]   (q with seqlen=T, k with seqlen=S)
// mode 2  : out bf16 [(b*H+h), HD, seqlen]   (v transposed for PV B-frags)
// mode 3  : X is ctx fp32 [B*T, D]; out fp32 [T, B, D]
__global__ __launch_bounds__(256, 1)
void gemm_bf16_kernel(const float* __restrict__ X, const float* __restrict__ Xadd,
                      const float* __restrict__ W, const float* __restrict__ bias,
                      float scale, int seqlen, void* __restrict__ outp, int mode)
{
    __shared__ __bf16 As[BM * LDT];
    __shared__ __bf16 Bs[BN * LDT];

    const int tid   = threadIdx.x;
    const int wave  = tid >> 5, lane = tid & 31;
    const int laneN = lane & 15, hi = lane >> 4;
    const int waveM = wave & 3,  waveN = wave >> 2;        // 4x2 waves -> 32x64 per wave
    const int blockN = blockIdx.x * BN;
    const int blockM = blockIdx.y * BM;

    const int lrow = tid & 127;        // loader: 1 row, 16 cols per thread
    const int lcg  = (tid >> 7) * 16;

    const int am = blockM + lrow;
    long aoff;
    if (mode == 3) {
        aoff = (long)am * D_DIM;
    } else {  // inputs are [seq, B, D]; row m = b*seqlen + sq
        const int bb = (am >= seqlen) ? 1 : 0;
        const int sq = am - (bb ? seqlen : 0);
        aoff = ((long)sq * BATCH + bb) * D_DIM;
    }
    const long boff = (long)(blockN + lrow) * D_DIM;  // W row n is contiguous (W is [out,in])

    v8f acc[2][4];
    const v8f zero8 = {0.f,0.f,0.f,0.f,0.f,0.f,0.f,0.f};
#pragma unroll
    for (int i = 0; i < 2; ++i)
#pragma unroll
        for (int j = 0; j < 4; ++j) acc[i][j] = zero8;

    for (int k0 = 0; k0 < D_DIM; k0 += BK) {
        // ---- stage A tile (fp32 -> bf16) ----
        {
            const float4* ap = reinterpret_cast<const float4*>(X + aoff + k0 + lcg);
            float4 f0 = ap[0], f1 = ap[1], f2 = ap[2], f3 = ap[3];
            if (Xadd) {
                const float4* pp = reinterpret_cast<const float4*>(Xadd + aoff + k0 + lcg);
                float4 p0 = pp[0], p1 = pp[1], p2 = pp[2], p3 = pp[3];
                f0.x+=p0.x; f0.y+=p0.y; f0.z+=p0.z; f0.w+=p0.w;
                f1.x+=p1.x; f1.y+=p1.y; f1.z+=p1.z; f1.w+=p1.w;
                f2.x+=p2.x; f2.y+=p2.y; f2.z+=p2.z; f2.w+=p2.w;
                f3.x+=p3.x; f3.y+=p3.y; f3.z+=p3.z; f3.w+=p3.w;
            }
            cvt_store16(&As[lrow * LDT + lcg], f0, f1, f2, f3);
        }
        // ---- stage B tile (weights) ----
        {
            const float4* wp = reinterpret_cast<const float4*>(W + boff + k0 + lcg);
            cvt_store16(&Bs[lrow * LDT + lcg], wp[0], wp[1], wp[2], wp[3]);
        }
        __syncthreads();

        // ---- WMMA compute: 2 M-frags x 4 N-frags per wave ----
        BF16x16 afr[2];
#pragma unroll
        for (int mf = 0; mf < 2; ++mf) {
            const int row = waveM * 32 + mf * 16 + laneN;
            const int off = row * LDT + (hi ? 16 : 0);          // A: 16 contiguous K per lane
            afr[mf].q[0] = *reinterpret_cast<const uint4*>(&As[off]);
            afr[mf].q[1] = *reinterpret_cast<const uint4*>(&As[off + 8]);
        }
#pragma unroll
        for (int nf = 0; nf < 4; ++nf) {
            const int col = waveN * 64 + nf * 16 + laneN;
            const int off = col * LDT + (hi ? 16 : 0);          // B: 16 contiguous K per lane
            BF16x16 bfr;
            bfr.q[0] = *reinterpret_cast<const uint4*>(&Bs[off]);
            bfr.q[1] = *reinterpret_cast<const uint4*>(&Bs[off + 8]);
#pragma unroll
            for (int mf = 0; mf < 2; ++mf)
                acc[mf][nf] = __builtin_amdgcn_wmma_f32_16x16x32_bf16(
                    false, afr[mf].v, false, bfr.v, (short)0, acc[mf][nf], false, false);
        }
        __syncthreads();
    }

    // ---- epilogue: C layout lane N = lane&15, M = 8*hi + vgpr ----
#pragma unroll
    for (int nf = 0; nf < 4; ++nf) {
        const int col  = blockN + waveN * 64 + nf * 16 + laneN;
        const float bv = bias ? bias[col] : 0.0f;
#pragma unroll
        for (int mf = 0; mf < 2; ++mf) {
            const int row0 = blockM + waveM * 32 + mf * 16 + (hi ? 8 : 0);
#pragma unroll
            for (int r = 0; r < 8; ++r) {
                const int   row = row0 + r;
                const float val = (acc[mf][nf][r] + bv) * scale;
                if (mode == 3) {
                    const int bb = row >> 10;           // row = b*T + t
                    const int t  = row & (T_LEN - 1);
                    reinterpret_cast<float*>(outp)[((long)t * BATCH + bb) * D_DIM + col] = val;
                } else {
                    const int bb = (row >= seqlen) ? 1 : 0;
                    const int sq = row - (bb ? seqlen : 0);
                    const int hh = col >> 6;
                    const int dd = col & 63;
                    long idx;
                    if (mode == 2)
                        idx = ((long)(bb * N_HEAD + hh) * HD + dd) * seqlen + sq;
                    else
                        idx = ((long)(bb * N_HEAD + hh) * seqlen + sq) * HD + dd;
                    reinterpret_cast<__bf16*>(outp)[idx] = (__bf16)val;
                }
            }
        }
    }
}

// Flash attention: one block = (bh, 128 query rows); 8 waves x 16 rows.
// q,k bf16 [BH, seq, HD]; vT bf16 [BH, HD, S]; mask fp32 [BH, T, S]; ctx fp32 [B*T, D].
constexpr int PLD = 136;   // per-wave P tile row stride (bf16), 272B -> 16B-aligned segs

__global__ __launch_bounds__(256, 1)
void flash_attn_kernel(const __bf16* __restrict__ qb, const __bf16* __restrict__ kb,
                       const __bf16* __restrict__ vT, const float* __restrict__ mask,
                       float* __restrict__ ctxo)
{
    __shared__ __bf16 pShared[8 * 16 * PLD];   // 34,816 B

    const int bh   = blockIdx.x;               // b*H + h
    const int tile = blockIdx.y;
    const int b = bh >> 4, h = bh & 15;
    const int tid = threadIdx.x, wave = tid >> 5, lane = tid & 31;
    const int laneN = lane & 15, hi = lane >> 4;
    const int abase = hi ? 16 : 0;             // A-frag K base
    const int mbase = hi ? 8 : 0;              // C-frag M base
    const int tRow0 = tile * 128 + wave * 16;

    // Q fragments (16 rows x 64 d = 2 A-frags), loaded once from global
    BF16x16 qf[2];
    {
        const __bf16* qrow = qb + ((long)bh * T_LEN + tRow0 + laneN) * HD;
#pragma unroll
        for (int kk = 0; kk < 2; ++kk) {
            qf[kk].q[0] = *reinterpret_cast<const uint4*>(qrow + kk * 32 + abase);
            qf[kk].q[1] = *reinterpret_cast<const uint4*>(qrow + kk * 32 + abase + 8);
        }
    }

    const v8f zero8 = {0.f,0.f,0.f,0.f,0.f,0.f,0.f,0.f};
    float mrow[8], lsum[8];
#pragma unroll
    for (int r = 0; r < 8; ++r) { mrow[r] = -1e30f; lsum[r] = 0.0f; }
    v8f cacc[4];
#pragma unroll
    for (int c = 0; c < 4; ++c) cacc[c] = zero8;

    __bf16* pW = pShared + wave * 16 * PLD;    // per-wave private -> no block barrier needed
    const float* mbh = mask + (long)bh * T_LEN * S_LEN;

    for (int sc = 0; sc < S_LEN; sc += 128) {
        // ---- prefetch next S-chunk of the mask stream (gfx1250 global_prefetch) ----
        if (sc + 128 < S_LEN) {
#pragma unroll
            for (int r = 0; r < 8; ++r) {
                const float* pf = &mbh[(long)(tRow0 + mbase + r) * S_LEN + sc + 128 + lane * 4];
                __builtin_prefetch(pf, 0, 1);
            }
        }
        // ---- scores: Q[16,64] @ K^T[64,128] ----
        v8f sacc[8];
#pragma unroll
        for (int nf = 0; nf < 8; ++nf) sacc[nf] = zero8;
#pragma unroll
        for (int kk = 0; kk < 2; ++kk) {
#pragma unroll
            for (int nf = 0; nf < 8; ++nf) {
                const __bf16* kp = kb + ((long)bh * S_LEN + sc + nf * 16 + laneN) * HD
                                 + kk * 32 + (hi ? 16 : 0);
                BF16x16 bfr;
                bfr.q[0] = *reinterpret_cast<const uint4*>(kp);
                bfr.q[1] = *reinterpret_cast<const uint4*>(kp + 8);
                sacc[nf] = __builtin_amdgcn_wmma_f32_16x16x32_bf16(
                    false, qf[kk].v, false, bfr.v, (short)0, sacc[nf], false, false);
            }
        }
        // ---- add mask: 512MB one-shot stream -> non-temporal loads (don't thrash L2) ----
#pragma unroll
        for (int nf = 0; nf < 8; ++nf) {
#pragma unroll
            for (int r = 0; r < 8; ++r) {
                const float mval = __builtin_nontemporal_load(
                    &mbh[(long)(tRow0 + mbase + r) * S_LEN + sc + nf * 16 + laneN]);
                sacc[nf][r] += mval;
            }
        }
        // ---- online softmax per row (16-lane half-group reductions) ----
#pragma unroll
        for (int r = 0; r < 8; ++r) {
            float mx = sacc[0][r];
#pragma unroll
            for (int nf = 1; nf < 8; ++nf) mx = fmaxf(mx, sacc[nf][r]);
#pragma unroll
            for (int off = 8; off > 0; off >>= 1) mx = fmaxf(mx, __shfl_xor(mx, off, 32));
            const float mnew  = fmaxf(mrow[r], mx);
            const float alpha = __expf(mrow[r] - mnew);
            mrow[r] = mnew;
            float rs = 0.0f;
#pragma unroll
            for (int nf = 0; nf < 8; ++nf) {
                const float p = __expf(sacc[nf][r] - mnew);
                sacc[nf][r] = p;
                rs += p;
            }
#pragma unroll
            for (int off = 8; off > 0; off >>= 1) rs += __shfl_xor(rs, off, 32);
            lsum[r] = lsum[r] * alpha + rs;
#pragma unroll
            for (int c = 0; c < 4; ++c) cacc[c][r] *= alpha;
            // spill P row to per-wave LDS in A-frag-consumable layout
#pragma unroll
            for (int nf = 0; nf < 8; ++nf)
                pW[(mbase + r) * PLD + nf * 16 + laneN] = (__bf16)sacc[nf][r];
        }
        // ---- ctx += P[16,128] @ V[128,64] (vT gives contiguous B-frags) ----
#pragma unroll
        for (int kk2 = 0; kk2 < 4; ++kk2) {
            BF16x16 afr;
            const int ao = laneN * PLD + kk2 * 32 + abase;
            afr.q[0] = *reinterpret_cast<const uint4*>(&pW[ao]);
            afr.q[1] = *reinterpret_cast<const uint4*>(&pW[ao + 8]);
#pragma unroll
            for (int nf2 = 0; nf2 < 4; ++nf2) {
                const __bf16* vp = vT + ((long)bh * HD + nf2 * 16 + laneN) * S_LEN
                                 + sc + kk2 * 32 + (hi ? 16 : 0);
                BF16x16 bfr;
                bfr.q[0] = *reinterpret_cast<const uint4*>(vp);
                bfr.q[1] = *reinterpret_cast<const uint4*>(vp + 8);
                cacc[nf2] = __builtin_amdgcn_wmma_f32_16x16x32_bf16(
                    false, afr.v, false, bfr.v, (short)0, cacc[nf2], false, false);
            }
        }
    }

    // ---- finalize: divide by l, scatter to ctx [B*T, D] ----
#pragma unroll
    for (int r = 0; r < 8; ++r) {
        const float inv = 1.0f / lsum[r];
        const int t = tRow0 + mbase + r;
#pragma unroll
        for (int c = 0; c < 4; ++c) {
            ctxo[((long)b * T_LEN + t) * D_DIM + h * HD + c * 16 + laneN] = cacc[c][r] * inv;
        }
    }
}

extern "C" void kernel_launch(void* const* d_in, const int* in_sizes, int n_in,
                              void* d_out, int out_size, void* d_ws, size_t ws_size,
                              hipStream_t stream) {
    (void)in_sizes; (void)n_in; (void)out_size; (void)ws_size;
    const float* hs    = (const float*)d_in[0];   // [T,B,D]
    const float* mask  = (const float*)d_in[1];   // [B*H,T,S]
    const float* pos   = (const float*)d_in[2];   // [T,B,D]
    const float* kvs   = (const float*)d_in[3];   // [S,B,D]
    const float* kvpos = (const float*)d_in[4];   // [S,B,D]
    const float* Wq = (const float*)d_in[5];  const float* bq = (const float*)d_in[6];
    const float* Wk = (const float*)d_in[7];  const float* bk = (const float*)d_in[8];
    const float* Wv = (const float*)d_in[9];  const float* bv = (const float*)d_in[10];
    const float* Wo = (const float*)d_in[11]; const float* bo = (const float*)d_in[12];

    // workspace layout (bf16 q/k/vT + fp32 ctx), ~44 MB
    __bf16* qbuf = reinterpret_cast<__bf16*>(d_ws);
    __bf16* kbuf = qbuf + (long)BATCH * N_HEAD * T_LEN * HD;   // +2M elems
    __bf16* vbuf = kbuf + (long)BATCH * N_HEAD * S_LEN * HD;   // +8M elems
    float*  ctx  = reinterpret_cast<float*>(vbuf + (long)BATCH * N_HEAD * S_LEN * HD);

    const float scaling = 0.125f;   // hd^-0.5 = 64^-0.5

    dim3 blk(256);
    // q = ((hs+pos) @ Wq^T + bq) * scaling   -> bf16 [BH, T, HD]
    gemm_bf16_kernel<<<dim3(D_DIM / BN, (BATCH * T_LEN) / BM), blk, 0, stream>>>(
        hs, pos, Wq, bq, scaling, T_LEN, qbuf, 0);
    // k = (kv+kvpos) @ Wk^T + bk             -> bf16 [BH, S, HD]
    gemm_bf16_kernel<<<dim3(D_DIM / BN, (BATCH * S_LEN) / BM), blk, 0, stream>>>(
        kvs, kvpos, Wk, bk, 1.0f, S_LEN, kbuf, 1);
    // v = kv @ Wv^T + bv                     -> bf16 [BH, HD, S] (transposed)
    gemm_bf16_kernel<<<dim3(D_DIM / BN, (BATCH * S_LEN) / BM), blk, 0, stream>>>(
        kvs, nullptr, Wv, bv, 1.0f, S_LEN, vbuf, 2);
    // flash attention -> ctx fp32 [B*T, D]
    flash_attn_kernel<<<dim3(BATCH * N_HEAD, T_LEN / 128), blk, 0, stream>>>(
        qbuf, kbuf, vbuf, mask, ctx);
    // out = ctx @ Wo^T + bo                  -> fp32 [T, B, D]
    gemm_bf16_kernel<<<dim3(D_DIM / BN, (BATCH * T_LEN) / BM), blk, 0, stream>>>(
        ctx, nullptr, Wo, bo, 1.0f, T_LEN, d_out, 3);
}
